// CrossAttentionWithDistBias_7017976562313
// MI455X (gfx1250) — matrix operand order repaired and checked by reading, not compiled, not run on hardware
//
#include <hip/hip_runtime.h>
#include <math.h>

// ---------------------------------------------------------------------------
// CDNA5 (gfx1250) cross-attention with distance bias.
// wave32, WMMA f32_16x16x32_bf16 for all GEMMs, fp32 softmax/RBF,
// async global->LDS staging of V^T, global prefetch for bias tiles.
// ---------------------------------------------------------------------------

typedef __bf16 bf16_t;
typedef __attribute__((ext_vector_type(16))) __bf16 v16bf;
typedef __attribute__((ext_vector_type(8)))  __bf16 v8bf;
typedef __attribute__((ext_vector_type(8)))  float  v8f;
typedef __attribute__((ext_vector_type(4)))  float  v4f;

#define D_MODEL 256
#define NHEAD   8
#define D_HEAD  32
#define B_SZ    2
#define TR      2048
#define TL      1024
#define NUM_RBF 16
#define SCALE_QK 0.17677669529663689f   // 32^-0.5
#define BIAS_SCALE 2.0f

__device__ __forceinline__ v8f zero8() {
  v8f z;
#pragma unroll
  for (int i = 0; i < 8; ++i) z[i] = 0.0f;
  return z;
}

__device__ __forceinline__ v8f wmma_bf16(v16bf a, v16bf b, v8f c) {
  // D = A(16x32) * B(32x16) + C, fp32 accumulate
  return __builtin_amdgcn_wmma_f32_16x16x32_bf16(
      /*neg_a=*/false, a, /*neg_b=*/false, b,
      /*c_mod=*/(short)0, c, /*reuse_a=*/false, /*reuse_b=*/false);
}

// A-fragment (16x32 bf16, MxK). ISA layout: lanes 0-15 -> M=lane, halves hold
// K = {8g..8g+7, 16+8g..16+8g+7} where g = lane>>4. Source row-major, `stride`
// elements between rows, base points at [row0][k0].
__device__ __forceinline__ v16bf load_A_bf16(const bf16_t* base, int stride, int lane) {
  const int m = lane & 15, g = lane >> 4;
  const bf16_t* r = base + (size_t)m * stride;
  v8bf lo = *(const v8bf*)(r + 8 * g);
  v8bf hi = *(const v8bf*)(r + 16 + 8 * g);
  v16bf a;
#pragma unroll
  for (int j = 0; j < 8; ++j) { a[j] = lo[j]; a[j + 8] = hi[j]; }
  return a;
}

// Same A-fragment but converting from fp32 source on the fly.
__device__ __forceinline__ v16bf load_A_f32(const float* base, int stride, int lane) {
  const int m = lane & 15, g = lane >> 4;
  const float* r = base + (size_t)m * stride;
  v4f a0 = *(const v4f*)(r + 8 * g);
  v4f a1 = *(const v4f*)(r + 8 * g + 4);
  v4f b0 = *(const v4f*)(r + 16 + 8 * g);
  v4f b1 = *(const v4f*)(r + 16 + 8 * g + 4);
  v16bf a;
#pragma unroll
  for (int j = 0; j < 4; ++j) {
    a[j]      = (__bf16)a0[j];
    a[j + 4]  = (__bf16)a1[j];
    a[j + 8]  = (__bf16)b0[j];
    a[j + 12] = (__bf16)b1[j];
  }
  return a;
}

// B-fragment (32x16 bf16, KxN). ISA layout: lane n holds column N=n, halves
// hold K = {8g..8g+7, 16+8g..16+8g+7}. Source is B^T row-major (lane n reads row n,
// two runs of 8 K values): base points at [n0][k0], `stride` elems per row.
__device__ __forceinline__ v16bf load_B_bf16(const bf16_t* base, int stride, int lane) {
  const int n = lane & 15, g = lane >> 4;
  const bf16_t* r = base + (size_t)n * stride + 8 * g;
  v8bf lo = *(const v8bf*)(r);
  v8bf hi = *(const v8bf*)(r + 16);
  v16bf b;
#pragma unroll
  for (int j = 0; j < 8; ++j) { b[j] = lo[j]; b[j + 8] = hi[j]; }
  return b;
}

// ---------------------------------------------------------------------------
// Kernel 1: transpose + bf16-convert the four weight matrices.
// dst[n*256+k] = (bf16) src[k*256+n]   (so WMMA B-frags read W^T rows)
// ---------------------------------------------------------------------------
__global__ void cvt_weights_kernel(const float* __restrict__ Wq,
                                   const float* __restrict__ Wk,
                                   const float* __restrict__ Wv,
                                   const float* __restrict__ Wo,
                                   bf16_t* __restrict__ wbase) {
  const float* srcs[4] = {Wq, Wk, Wv, Wo};
  const float* src = srcs[blockIdx.y];
  bf16_t* dst = wbase + (size_t)blockIdx.y * (D_MODEL * D_MODEL);
  int idx = blockIdx.x * 256 + threadIdx.x;   // 0..65535
  int n = idx >> 8, k = idx & 255;
  dst[(size_t)n * 256 + k] = (__bf16)src[(size_t)k * 256 + n];
}

// ---------------------------------------------------------------------------
// Kernel 2: projection GEMM  Y = X @ W + b  via WMMA, K=256 in 8 chunks of 32.
// mode 0/1: store (b,h,token,d) bf16 (Q / K).  mode 2: store V transposed
// (b,h,d,token) so attn@V B-fragments are contiguous loads.
// grid.x = (B*Ttok)/16, block = 256 (8 waves; wave w owns out cols 32w..32w+31)
// ---------------------------------------------------------------------------
__global__ void __launch_bounds__(256)
proj_kernel(const float* __restrict__ X, const bf16_t* __restrict__ WT,
            const float* __restrict__ bias, bf16_t* __restrict__ dst,
            int Ttok, int mode) {
  const int lane = threadIdx.x & 31;
  const int w = threadIdx.x >> 5;
  const int g = lane >> 4;
  const int row0 = blockIdx.x * 16;          // global token row in [0, B*Ttok)
  const int b = row0 / Ttok;
  const int r0 = row0 - b * Ttok;
  const int n0 = w * 32;

  v8f c0 = zero8(), c1 = zero8();
#pragma unroll
  for (int kc = 0; kc < 8; ++kc) {
    v16bf a  = load_A_f32(X + (size_t)row0 * D_MODEL + kc * 32, D_MODEL, lane);
    v16bf b0 = load_B_bf16(WT + (size_t)n0 * D_MODEL + kc * 32, D_MODEL, lane);
    v16bf b1 = load_B_bf16(WT + (size_t)(n0 + 16) * D_MODEL + kc * 32, D_MODEL, lane);
    c0 = wmma_bf16(a, b0, c0);
    c1 = wmma_bf16(a, b1, c1);
  }
  const float bb0 = bias[n0 + (lane & 15)];
  const float bb1 = bias[n0 + 16 + (lane & 15)];
  // head index: both 16-col tiles of this wave live in head h = w
  if (mode < 2) {
    bf16_t* d0 = dst + (size_t)(b * NHEAD + w) * Ttok * D_HEAD;
#pragma unroll
    for (int i = 0; i < 8; ++i) {
      int r = r0 + i + 8 * g;
      d0[(size_t)r * D_HEAD + (lane & 15)]      = (__bf16)(c0[i] + bb0);
      d0[(size_t)r * D_HEAD + 16 + (lane & 15)] = (__bf16)(c1[i] + bb1);
    }
  } else {
    bf16_t* d0 = dst + (size_t)(b * NHEAD + w) * D_HEAD * TL;
#pragma unroll
    for (int i = 0; i < 8; ++i) {
      int l = r0 + i + 8 * g;
      d0[(size_t)(lane & 15) * TL + l]        = (__bf16)(c0[i] + bb0);
      d0[(size_t)(16 + (lane & 15)) * TL + l] = (__bf16)(c1[i] + bb1);
    }
  }
}

// ---------------------------------------------------------------------------
// Kernel 3: RBF distance bias, computed ONCE per (b,r,l), broadcast to all 8
// heads, scaled by BIAS_SCALE, mask folded in as -inf. Written into the attn
// output buffer (B,H,Tr,Tl); the attention kernel reads then overwrites it.
// grid = (TL/256, TR, B)
// ---------------------------------------------------------------------------
__global__ void __launch_bounds__(256)
bias_kernel(const float* __restrict__ rec_c, const float* __restrict__ lig_c,
            const unsigned char* __restrict__ mask,
            const float* __restrict__ W_rbf, const float* __restrict__ mus,
            const float* __restrict__ gammap, float* __restrict__ attn) {
  const int l = blockIdx.x * 256 + threadIdx.x;
  const int r = blockIdx.y;
  const int b = blockIdx.z;
  const float gamma = gammap[0];
  const float rx = rec_c[(size_t)(b * TR + r) * 3 + 0];
  const float ry = rec_c[(size_t)(b * TR + r) * 3 + 1];
  const float rz = rec_c[(size_t)(b * TR + r) * 3 + 2];
  const float lx = lig_c[(size_t)(b * TL + l) * 3 + 0];
  const float ly = lig_c[(size_t)(b * TL + l) * 3 + 1];
  const float lz = lig_c[(size_t)(b * TL + l) * 3 + 2];
  const float dx = rx - lx, dy = ry - ly, dz = rz - lz;
  const float d = sqrtf(fmaxf(dx * dx + dy * dy + dz * dz, 0.0f));
  float e[NUM_RBF];
#pragma unroll
  for (int n = 0; n < NUM_RBF; ++n) {
    float t = d - mus[n];
    e[n] = __expf(-gamma * t * t);
  }
  const bool msk = mask[(size_t)b * TL + l] != 0;
#pragma unroll
  for (int h = 0; h < NHEAD; ++h) {
    float acc = 0.0f;
#pragma unroll
    for (int n = 0; n < NUM_RBF; ++n) acc = fmaf(e[n], W_rbf[n * NHEAD + h], acc);
    float v = msk ? -__builtin_inff() : BIAS_SCALE * acc;
    attn[((size_t)(b * NHEAD + h) * TR + r) * TL + l] = v;
  }
}

// ---------------------------------------------------------------------------
// Kernel 4: fused attention for one (b, h, 16-row tile).
// 8 waves; wave w owns ligand columns [128w, 128w+128).
//   - V^T (b,h) slice (contiguous 64KB) is staged global->LDS with async
//     copies issued at kernel entry; completion waited (s_wait_asynccnt)
//     only after the softmax phase, so the DMA is fully hidden.
//   - logits = WMMA(Q, K^T)*scale + bias (bias buffer prefetched, already
//     carries mask(-inf) and BIAS_SCALE), workgroup softmax via LDS slots,
//     attn written fp32, probs staged bf16 in LDS as WMMA A-frags vs V^T.
// grid = (TR/16, NHEAD, B)
// ---------------------------------------------------------------------------
#define V_ASYNC_CP(OFF)                                                        \
  asm volatile("global_load_async_to_lds_b128 %0, %1, off offset:" #OFF        \
               :: "v"(ldsb), "v"(gsrc) : "memory")

__global__ void __launch_bounds__(256)
attn_kernel(const bf16_t* __restrict__ qb, const bf16_t* __restrict__ kb,
            const bf16_t* __restrict__ vt, float* __restrict__ attn,
            bf16_t* __restrict__ outp) {
  __shared__ __align__(16) bf16_t vbuf[32][TL];       // 64 KB: staged V^T slice
  __shared__ __align__(16) bf16_t pbuf[8][16][128];   // 32 KB: unnormalized probs
  __shared__ float wmax[8][16];
  __shared__ float wsum[8][16];
  float* outw = (float*)&pbuf[0][0][0];               // 16 KB alias (pbuf dead by then)

  const int lane = threadIdx.x & 31;
  const int w = threadIdx.x >> 5;
  const int g = lane >> 4;
  const int r0 = blockIdx.x * 16;
  const int h = blockIdx.y, b = blockIdx.z;
  const int bh = b * NHEAD + h;
  const int c0 = w * 128;

  const bf16_t* Q = qb + ((size_t)bh * TR + r0) * D_HEAD;
  const bf16_t* K = kb + (size_t)bh * TL * D_HEAD;
  const bf16_t* V = vt + (size_t)bh * D_HEAD * TL;
  float* attnBH = attn + ((size_t)bh * TR + r0) * TL;

  // ---- async-stage V^T slice (flat 64KB; wave w copies bytes [8K*w, 8K*w+8K))
  {
    unsigned ldsb = (unsigned)(size_t)(&vbuf[0][0]) + (unsigned)(w * 8192 + lane * 16);
    const char* gsrc = (const char*)V + w * 8192 + lane * 16;
    V_ASYNC_CP(0);    V_ASYNC_CP(512);  V_ASYNC_CP(1024); V_ASYNC_CP(1536);
    V_ASYNC_CP(2048); V_ASYNC_CP(2560); V_ASYNC_CP(3072); V_ASYNC_CP(3584);
    V_ASYNC_CP(4096); V_ASYNC_CP(4608); V_ASYNC_CP(5120); V_ASYNC_CP(5632);
    V_ASYNC_CP(6144); V_ASYNC_CP(6656); V_ASYNC_CP(7168); V_ASYNC_CP(7680);
  }

  // ---- prefetch this wave's bias tile (16 rows x 512B, 128B granularity) ----
  {
    const char* bp = (const char*)(attnBH + (size_t)(lane & 15) * TL + c0) + g * 256;
    __builtin_prefetch(bp, 0, 3);
    __builtin_prefetch(bp + 128, 0, 3);
  }

  // ---- logits: one WMMA per 16x16 tile (K dim = D_HEAD = 32) ----
  v16bf aq = load_A_bf16(Q, D_HEAD, lane);
  v8f c[8];
#pragma unroll
  for (int t = 0; t < 8; ++t) {
    v16bf bk = load_B_bf16(K + (size_t)(c0 + 16 * t) * D_HEAD, D_HEAD, lane);
    c[t] = wmma_bf16(aq, bk, zero8());
  }
  // ---- scale + bias (bias buffer already holds mask(-inf) and 2*bias) ----
#pragma unroll
  for (int t = 0; t < 8; ++t)
#pragma unroll
    for (int i = 0; i < 8; ++i) {
      float bv = attnBH[(size_t)(i + 8 * g) * TL + c0 + 16 * t + (lane & 15)];
      c[t][i] = c[t][i] * SCALE_QK + bv;
    }

  // ---- row max: in-thread, cross-lane (16-wide), cross-wave via LDS ----
  float mx[8];
#pragma unroll
  for (int i = 0; i < 8; ++i) {
    float m = c[0][i];
#pragma unroll
    for (int t = 1; t < 8; ++t) m = fmaxf(m, c[t][i]);
#pragma unroll
    for (int off = 8; off >= 1; off >>= 1) m = fmaxf(m, __shfl_xor(m, off, 32));
    mx[i] = m;
  }
  if ((lane & 15) == 0) {
#pragma unroll
    for (int i = 0; i < 8; ++i) wmax[w][i + 8 * g] = mx[i];
  }
  __syncthreads();
#pragma unroll
  for (int i = 0; i < 8; ++i) {
    float m = wmax[0][i + 8 * g];
#pragma unroll
    for (int ww = 1; ww < 8; ++ww) m = fmaxf(m, wmax[ww][i + 8 * g]);
    mx[i] = m;
  }

  // ---- exp, row sums, stage unnormalized probs into LDS as bf16 ----
  float sm[8];
#pragma unroll
  for (int i = 0; i < 8; ++i) sm[i] = 0.0f;
#pragma unroll
  for (int t = 0; t < 8; ++t)
#pragma unroll
    for (int i = 0; i < 8; ++i) {
      float p = __expf(c[t][i] - mx[i]);
      c[t][i] = p;
      sm[i] += p;
      pbuf[w][i + 8 * g][16 * t + (lane & 15)] = (__bf16)p;
    }
#pragma unroll
  for (int i = 0; i < 8; ++i) {
#pragma unroll
    for (int off = 8; off >= 1; off >>= 1) sm[i] += __shfl_xor(sm[i], off, 32);
  }
  if ((lane & 15) == 0) {
#pragma unroll
    for (int i = 0; i < 8; ++i) wsum[w][i + 8 * g] = sm[i];
  }
  // all of this wave's async V copies must be in LDS before anyone reads vbuf
  asm volatile("s_wait_asynccnt 0x0" ::: "memory");
  __syncthreads();
  float inv[8];
#pragma unroll
  for (int i = 0; i < 8; ++i) {
    float s = wsum[0][i + 8 * g];
#pragma unroll
    for (int ww = 1; ww < 8; ++ww) s += wsum[ww][i + 8 * g];
    inv[i] = 1.0f / s;
  }

  // ---- write normalized attention probabilities (fp32) ----
#pragma unroll
  for (int t = 0; t < 8; ++t)
#pragma unroll
    for (int i = 0; i < 8; ++i)
      attnBH[(size_t)(i + 8 * g) * TL + c0 + 16 * t + (lane & 15)] = c[t][i] * inv[i];

  // ---- attn @ V from LDS (4 WMMA k-steps x 2 d-tiles) ----
  v8f o0 = zero8(), o1 = zero8();
#pragma unroll
  for (int kc = 0; kc < 4; ++kc) {
    v16bf ap = load_A_bf16(&pbuf[w][0][32 * kc], 128, lane);
    v16bf b0 = load_B_bf16(&vbuf[0][c0 + 32 * kc], TL, lane);
    v16bf b1 = load_B_bf16(&vbuf[16][c0 + 32 * kc], TL, lane);
    o0 = wmma_bf16(ap, b0, o0);
    o1 = wmma_bf16(ap, b1, o1);
  }
  __syncthreads();   // pbuf reads done everywhere; outw aliases pbuf below

  // row-normalize the partial (pbuf held unnormalized probs), park in LDS slot
#pragma unroll
  for (int i = 0; i < 8; ++i) {
    outw[w * 512 + (i + 8 * g) * 32 + (lane & 15)]      = o0[i] * inv[i];
    outw[w * 512 + (i + 8 * g) * 32 + 16 + (lane & 15)] = o1[i] * inv[i];
  }
  __syncthreads();

  // ---- cross-wave reduction, store out' (b, r, h*32+d) as bf16 ----
  for (int e = threadIdx.x; e < 512; e += 256) {
    float s = outw[e];
#pragma unroll
    for (int ww = 1; ww < 8; ++ww) s += outw[ww * 512 + e];
    int r = e >> 5, d = e & 31;
    outp[((size_t)(b * TR) + r0 + r) * D_MODEL + h * D_HEAD + d] = (__bf16)s;
  }
}

// ---------------------------------------------------------------------------
// Kernel 5: out = out' @ Wo + bo, fp32 result. grid.x = (B*TR)/16, block 256.
// ---------------------------------------------------------------------------
__global__ void __launch_bounds__(256)
outproj_kernel(const bf16_t* __restrict__ outp, const bf16_t* __restrict__ WoT,
               const float* __restrict__ bo, float* __restrict__ out) {
  const int lane = threadIdx.x & 31;
  const int w = threadIdx.x >> 5;
  const int g = lane >> 4;
  const int row0 = blockIdx.x * 16;
  const int n0 = w * 32;
  v8f c0 = zero8(), c1 = zero8();
#pragma unroll
  for (int kc = 0; kc < 8; ++kc) {
    v16bf a  = load_A_bf16(outp + (size_t)row0 * D_MODEL + kc * 32, D_MODEL, lane);
    v16bf b0 = load_B_bf16(WoT + (size_t)n0 * D_MODEL + kc * 32, D_MODEL, lane);
    v16bf b1 = load_B_bf16(WoT + (size_t)(n0 + 16) * D_MODEL + kc * 32, D_MODEL, lane);
    c0 = wmma_bf16(a, b0, c0);
    c1 = wmma_bf16(a, b1, c1);
  }
  const float bb0 = bo[n0 + (lane & 15)];
  const float bb1 = bo[n0 + 16 + (lane & 15)];
#pragma unroll
  for (int i = 0; i < 8; ++i) {
    out[(size_t)(row0 + i + 8 * g) * D_MODEL + n0 + (lane & 15)]      = c0[i] + bb0;
    out[(size_t)(row0 + i + 8 * g) * D_MODEL + n0 + 16 + (lane & 15)] = c1[i] + bb1;
  }
}

// ---------------------------------------------------------------------------
extern "C" void kernel_launch(void* const* d_in, const int* in_sizes, int n_in,
                              void* d_out, int out_size, void* d_ws, size_t ws_size,
                              hipStream_t stream) {
  const float* rec_tokens  = (const float*)d_in[0];
  const float* lig_tokens  = (const float*)d_in[1];
  const float* rec_centers = (const float*)d_in[2];
  const float* lig_centers = (const float*)d_in[3];
  const unsigned char* lig_mask = (const unsigned char*)d_in[4];
  const float* Wq = (const float*)d_in[5];
  const float* bq = (const float*)d_in[6];
  const float* Wk = (const float*)d_in[7];
  const float* bk = (const float*)d_in[8];
  const float* Wv = (const float*)d_in[9];
  const float* bv = (const float*)d_in[10];
  const float* Wo = (const float*)d_in[11];
  const float* bo = (const float*)d_in[12];
  const float* W_rbf = (const float*)d_in[13];
  const float* mus   = (const float*)d_in[14];
  const float* gamma = (const float*)d_in[15];

  float* out  = (float*)d_out;                               // (B,Tr,256)
  float* attn = out + (size_t)B_SZ * TR * D_MODEL;           // (B,H,Tr,Tl)

  // workspace layout (bf16)
  bf16_t* WqT = (bf16_t*)d_ws;                               // 4 x 256x256
  bf16_t* WkT = WqT + (size_t)D_MODEL * D_MODEL;
  bf16_t* WvT = WkT + (size_t)D_MODEL * D_MODEL;
  bf16_t* WoT = WvT + (size_t)D_MODEL * D_MODEL;
  bf16_t* qb  = WoT + (size_t)D_MODEL * D_MODEL;             // (B,H,Tr,32)
  bf16_t* kb  = qb + (size_t)B_SZ * NHEAD * TR * D_HEAD;     // (B,H,Tl,32)
  bf16_t* vt  = kb + (size_t)B_SZ * NHEAD * TL * D_HEAD;     // (B,H,32,Tl)
  bf16_t* outp = vt + (size_t)B_SZ * NHEAD * TL * D_HEAD;    // (B,Tr,256)

  cvt_weights_kernel<<<dim3(256, 4), 256, 0, stream>>>(Wq, Wk, Wv, Wo, WqT);
  proj_kernel<<<dim3((B_SZ * TR) / 16), 256, 0, stream>>>(rec_tokens, WqT, bq, qb, TR, 0);
  proj_kernel<<<dim3((B_SZ * TL) / 16), 256, 0, stream>>>(lig_tokens, WkT, bk, kb, TL, 1);
  proj_kernel<<<dim3((B_SZ * TL) / 16), 256, 0, stream>>>(lig_tokens, WvT, bv, vt, TL, 2);
  bias_kernel<<<dim3(TL / 256, TR, B_SZ), 256, 0, stream>>>(
      rec_centers, lig_centers, lig_mask, W_rbf, mus, gamma, attn);
  attn_kernel<<<dim3(TR / 16, NHEAD, B_SZ), 256, 0, stream>>>(qb, kb, vt, attn, outp);
  outproj_kernel<<<dim3((B_SZ * TR) / 16), 256, 0, stream>>>(outp, WoT, bo, out);
}
